// MobileNetV2MEInt_63367947485646
// MI455X (gfx1250) — matrix operand order repaired and checked by reading, not compiled, not run on hardware
//
#include <hip/hip_runtime.h>

// ---------------- problem constants ----------------
#define N_PTS 200000
#define CIN   96
#define CHID  384
#define KVOL  9

typedef __attribute__((ext_vector_type(16))) _Float16 v16h;
typedef __attribute__((ext_vector_type(8)))  float    v8f;

// ---------------- workspace layout (in _Float16 units) ----------------
// [W1 swizzled B-fragments][W3 swizzled B-fragments][h1 (N x CHID) f16]
#define W1SW_HALFS (24 * 3 * 32 * 16)   // 24 col-tiles, 3 k-chunks of 32
#define W3SW_HALFS (6 * 12 * 32 * 16)   // 6 col-tiles, 12 k-chunks of 32
#define H1_OFF     (W1SW_HALFS + W3SW_HALFS)

// CDNA5 WMMA f16 fragment K maps (ISA 7.12.2 / 7.12.5):
// A 16x32: lane holds M=lane%16; element e -> K = (e<8 ? e : e+8) + 8*(lane>=16)
__device__ __forceinline__ int kmapA(int e, int hi) {
  return (e < 8 ? e : e + 8) + (hi ? 8 : 0);
}
// B 32x16: lane holds N=lane%16; element e -> K = (e<8 ? e : e+8) + 8*(lane>=16)
__device__ __forceinline__ int kmapB(int e, int hi) {
  return (e < 8 ? e : e + 8) + (hi ? 8 : 0);
}

// ---------------- kernel 0: swizzle weights into B-fragment layout ----------------
__global__ void __launch_bounds__(256) prep_weights(
    const float* __restrict__ W1, const float* __restrict__ W3,
    _Float16* __restrict__ ws) {
  int gid = blockIdx.x * 256 + threadIdx.x;
  if (gid < W1SW_HALFS) {
    int e    = gid & 15;
    int lane = (gid >> 4) & 31;
    int cc   = (gid >> 9) % 3;          // k-chunk
    int t    = gid / (3 * 512);         // col tile
    int k    = cc * 32 + kmapB(e, lane >= 16);
    int col  = t * 16 + (lane & 15);
    ws[gid] = (_Float16)W1[k * CHID + col];       // W1 is (CIN=96, CHID=384)
  } else if (gid < W1SW_HALFS + W3SW_HALFS) {
    int g    = gid - W1SW_HALFS;
    int e    = g & 15;
    int lane = (g >> 4) & 31;
    int cc   = (g >> 9) % 12;
    int t    = g / (12 * 512);
    int k    = cc * 32 + kmapB(e, lane >= 16);
    int col  = t * 16 + (lane & 15);
    ws[W1SW_HALFS + g] = (_Float16)W3[k * CIN + col];  // W3 is (CHID=384, CIN=96)
  }
}

// ---------------- kernel 1: h1 = clamp_relu6(qscale(x@W1 + b1, s1)) -> f16 ----------------
__global__ void __launch_bounds__(256) gemm1_kernel(
    const float* __restrict__ x, const float* __restrict__ b1,
    const float* __restrict__ s1, const int* __restrict__ bs_p,
    _Float16* __restrict__ ws) {
  __shared__ _Float16 sA[3 * 32 * 16];   // A fragments: [kchunk][lane][e]

  const int tid = threadIdx.x;
  const int n0  = blockIdx.x * 16;
  const int bs  = *bs_p;
  const float inv2bs = 1.0f / (float)(1 << bs);
  const float qmax   = (float)(1 << (bs - 1));

  // stage x tile (16 x 96) into LDS directly in A-fragment layout
  for (int i = tid; i < 3 * 32 * 16; i += 256) {
    int e    = i & 15;
    int lane = (i >> 4) & 31;
    int cc   = i >> 9;
    int k    = cc * 32 + kmapA(e, lane >= 16);
    int row  = n0 + (lane & 15);
    sA[i] = (_Float16)x[row * CIN + k];
  }
  __syncthreads();

  const int lane = tid & 31;
  const int wave = tid >> 5;
  const v16h* sAv = (const v16h*)sA;
  const v16h* Bsw = (const v16h*)ws;                 // W1 fragments
  _Float16* h1 = ws + H1_OFF;

  for (int ct = wave; ct < 24; ct += 8) {            // 24 column tiles of 16
    v8f acc = {};
#pragma unroll
    for (int cc = 0; cc < 3; ++cc) {                 // K = 96 = 3 x 32
      v16h a = sAv[cc * 32 + lane];
      v16h b = Bsw[(ct * 3 + cc) * 32 + lane];
      acc = __builtin_amdgcn_wmma_f32_16x16x32_f16(false, a, false, b,
                                                   (short)0, acc, false, false);
    }
    int c = ct * 16 + (lane & 15);
    float bc = b1[c];
    float sc = rintf(s1[c]);                          // matches jnp.round (RNE)
#pragma unroll
    for (int r = 0; r < 8; ++r) {
      int n = n0 + r + ((lane >> 4) << 3);            // C/D row map
      float q = (acc[r] + bc) * sc * inv2bs;          // qscale
      q = fminf(fmaxf(q, -qmax), qmax);               // +-qmax clamp
      q = fminf(fmaxf(q, 0.0f), 6.0f);                // ReLU6
      h1[n * CHID + c] = (_Float16)q;
    }
  }
}

// ---------------- kernel 2: depthwise gather + GEMM2 + residual ----------------
__global__ void __launch_bounds__(256) dw_gemm2_kernel(
    const float* __restrict__ x,
    const float* __restrict__ Wdw, const float* __restrict__ bdw,
    const float* __restrict__ b3,  const float* __restrict__ s2,
    const float* __restrict__ s3,  const float* __restrict__ main_s,
    const float* __restrict__ res_s, const int* __restrict__ nbr,
    const int* __restrict__ bs_p,
    const _Float16* __restrict__ ws, float* __restrict__ out) {
  __shared__ float    sWdw[KVOL * CHID];   // 13.5 KB
  __shared__ float    sBdw[CHID];
  __shared__ float    sS2[CHID];
  __shared__ int      sNbr[16 * KVOL];
  __shared__ _Float16 sA2[12 * 32 * 16];   // h2 tile in A-fragment layout, 12 KB

  const int tid = threadIdx.x;
  const int n0  = blockIdx.x * 16;
  const int bs  = *bs_p;
  const float two_bs = (float)(1 << bs);
  const float inv2bs = 1.0f / two_bs;
  const float qmax   = (float)(1 << (bs - 1));

  for (int i = tid; i < KVOL * CHID; i += 256) sWdw[i] = Wdw[i];
  for (int i = tid; i < CHID; i += 256) { sBdw[i] = bdw[i]; sS2[i] = s2[i]; }
  for (int i = tid; i < 16 * KVOL; i += 256) sNbr[i] = nbr[n0 * KVOL + i];
  __syncthreads();

  // ---- depthwise 3x3 via kernel-map gather; 16 threads per row ----
  const int n  = tid >> 4;     // 0..15 (row in tile)
  const int c0 = tid & 15;     // channel phase; c = c0 + 16j, coalesced over lanes
  const _Float16* h1 = ws + H1_OFF;

  int m[KVOL];
#pragma unroll
  for (int k = 0; k < KVOL; ++k) m[k] = sNbr[n * KVOL + k];

  float acc[24];
#pragma unroll
  for (int j = 0; j < 24; ++j) acc[j] = 0.0f;

  for (int k = 0; k < KVOL; ++k) {
    const _Float16* hr = h1 + (size_t)m[k] * CHID;   // gathered row (mostly L2-hit: h1 fits L2)
    const float*    wr = sWdw + k * CHID;
#pragma unroll
    for (int j = 0; j < 24; ++j) {
      int c = c0 + (j << 4);
      acc[j] += (float)hr[c] * wr[c];
    }
  }

#pragma unroll
  for (int j = 0; j < 24; ++j) {
    int c = c0 + (j << 4);
    float q = (acc[j] + sBdw[c]) * rintf(sS2[c]) * inv2bs;
    q = fminf(fmaxf(q, -qmax), qmax);
    q = fminf(fmaxf(q, 0.0f), 6.0f);
    // write h2 into LDS in A-fragment layout (inverse of kmapA)
    int cc = c >> 5, kk = c & 31;
    int hi = (kk >> 3) & 1;
    int e  = (kk & 7) + (kk >= 16 ? 8 : 0);
    int fl = n + (hi << 4);
    sA2[(cc * 32 + fl) * 16 + e] = (_Float16)q;
  }
  __syncthreads();

  // ---- GEMM2: (16 x 384) @ (384 x 96), 6 column tiles, one per wave ----
  const int lane = tid & 31;
  const int wave = tid >> 5;
  if (wave < 6) {
    const v16h* sA2v = (const v16h*)sA2;
    const v16h* B3   = (const v16h*)(ws + W1SW_HALFS);  // W3 fragments
    v8f acc3 = {};
#pragma unroll
    for (int cc = 0; cc < 12; ++cc) {                   // K = 384 = 12 x 32
      v16h a = sA2v[cc * 32 + lane];
      v16h b = B3[(wave * 12 + cc) * 32 + lane];
      acc3 = __builtin_amdgcn_wmma_f32_16x16x32_f16(false, a, false, b,
                                                    (short)0, acc3, false, false);
    }
    int c = wave * 16 + (lane & 15);
    float bc = b3[c];
    float sc = rintf(s3[c]);
    float ms = rintf(main_s[0]) * two_bs;
    float rs = rintf(res_s[0])  * two_bs;
#pragma unroll
    for (int r = 0; r < 8; ++r) {
      int nn = n0 + r + ((lane >> 4) << 3);
      float q = (acc3[r] + bc) * sc * inv2bs;
      q = fminf(fmaxf(q, -qmax), qmax);                 // +-qmax clamp (no ReLU6 here)
      out[nn * CIN + c] = ms * q + rs * x[nn * CIN + c]; // scaled residual
    }
  }
}

// ---------------- launcher ----------------
extern "C" void kernel_launch(void* const* d_in, const int* in_sizes, int n_in,
                              void* d_out, int out_size, void* d_ws, size_t ws_size,
                              hipStream_t stream) {
  const float* x      = (const float*)d_in[0];
  const float* W1     = (const float*)d_in[1];
  const float* b1     = (const float*)d_in[2];
  const float* Wdw    = (const float*)d_in[3];
  const float* bdw    = (const float*)d_in[4];
  const float* W3     = (const float*)d_in[5];
  const float* b3     = (const float*)d_in[6];
  const float* s1     = (const float*)d_in[7];
  const float* s2     = (const float*)d_in[8];
  const float* s3     = (const float*)d_in[9];
  const float* main_s = (const float*)d_in[10];
  const float* res_s  = (const float*)d_in[11];
  const int*   nbr    = (const int*)d_in[12];
  const int*   bsP    = (const int*)d_in[13];

  _Float16* ws  = (_Float16*)d_ws;
  float*    out = (float*)d_out;

  const int swizTotal = W1SW_HALFS + W3SW_HALFS;
  prep_weights<<<(swizTotal + 255) / 256, 256, 0, stream>>>(W1, W3, ws);

  const int nBlocks = N_PTS / 16;   // 200000 / 16 = 12500 exactly
  gemm1_kernel<<<nBlocks, 256, 0, stream>>>(x, b1, s1, bsP, ws);
  dw_gemm2_kernel<<<nBlocks, 256, 0, stream>>>(x, Wdw, bdw, b3, s2, s3,
                                               main_s, res_s, nbr, bsP, ws, out);
}
